// Transformer_61314953118401
// MI455X (gfx1250) — matrix operand order repaired and checked by reading, not compiled, not run on hardware
//
#include <hip/hip_runtime.h>

// ---------------------------------------------------------------------------
// MI455X (gfx1250): bf16 WMMA for all matrix math, TDM async-load for LDS
// staging of the phase-2 S matrices.
// ---------------------------------------------------------------------------

typedef __bf16 bf16_t;
typedef __attribute__((ext_vector_type(16))) __bf16 v16bf;
typedef __attribute__((ext_vector_type(8)))  __bf16 v8bf;
typedef __attribute__((ext_vector_type(8)))  float  v8f;
typedef unsigned int u32x4 __attribute__((ext_vector_type(4)));
typedef int          i32x4 __attribute__((ext_vector_type(4)));
typedef int          i32x8 __attribute__((ext_vector_type(8)));

#define LCN 2500   // sequence length after convs
#define LKP 2528   // padded L for phase-1 K-chunks (79*32)
#define LQP 2560   // padded L for phase-2 row tiles (160*16)
#define NBH 128    // B*H = 16*8

__device__ __forceinline__ v8f wz8() {
  v8f z;
#pragma unroll
  for (int i = 0; i < 8; ++i) z[i] = 0.0f;
  return z;
}

__device__ __forceinline__ v16bf join16(v8bf lo, v8bf hi) {
  return __builtin_shufflevector(lo, hi, 0,1,2,3,4,5,6,7,8,9,10,11,12,13,14,15);
}

__device__ __forceinline__ v8f wmma_bf(v16bf a, v16bf b, v8f c) {
  // D = A(16x32 bf16) * B(32x16 bf16) + C(16x16 f32)
  return __builtin_amdgcn_wmma_f32_16x16x32_bf16(false, a, false, b, (short)0, c,
                                                 false, false);
}

// A-fragment (16x32, bf16, row-major source with leading dim ld):
// lane row = m0 + (lane&15); kpart = (lane>>4)*8; elems 0..7 -> k=kpart+e,
// elems 8..15 -> k=16+kpart+(e-8).   (ISA 7.12.2, 16-bit A 16x32)
__device__ __forceinline__ v16bf load_afrag(const bf16_t* __restrict__ base,
                                            int lane, size_t ld, int row0, int k0) {
  int row = row0 + (lane & 15);
  int kpart = (lane >> 4) * 8;
  const bf16_t* p = base + (size_t)row * ld + k0 + kpart;
  v8bf lo = *(const v8bf*)(p);
  v8bf hi = *(const v8bf*)(p + 16);
  return join16(lo, hi);
}

// ---------------------------------------------------------------------------
// Generic wave-per-tile WMMA GEMM: C(MxN) = A(MxK) * B(KxN) + bias (+epilogue)
// Requirements: M%16==0, N%128==0 per grid.y block of 8 waves, K%32==0.
// ---------------------------------------------------------------------------
template <bool RELU, bool RESID, bool OUTF, bool OUTB>
__global__ void gemm_wmma_k(const bf16_t* __restrict__ A,
                            const bf16_t* __restrict__ Bw,
                            const float* __restrict__ bias,
                            float* __restrict__ Cf,
                            bf16_t* __restrict__ Cb,
                            const float* __restrict__ Res,
                            int N, int K) {
  const int wave = threadIdx.x >> 5;
  const int lane = threadIdx.x & 31;
  const int m0 = blockIdx.x * 16;
  const int n0 = (blockIdx.y * 8 + wave) * 16;

  v8f acc = wz8();
  for (int kc = 0; kc < K; kc += 32) {
    v16bf a = load_afrag(A, lane, (size_t)K, m0, kc);
    // B fragment: lane holds column n0+(lane&15), k in table order.
    v16bf b;
#pragma unroll
    for (int e = 0; e < 16; ++e) {
      const int kk = kc + ((e < 8) ? ((lane >> 4) * 8 + e) : (16 + (lane >> 4) * 8 + e - 8));
      b[e] = Bw[(size_t)kk * N + n0 + (lane & 15)];
    }
    acc = wmma_bf(a, b, acc);
  }

  const int col = n0 + (lane & 15);
  const int rbase = m0 + ((lane >> 4) * 8);
  const float bv = bias[col];
#pragma unroll
  for (int r = 0; r < 8; ++r) {
    const size_t orow = (size_t)(rbase + r);
    float v = acc[r] + bv;
    if (RELU) v = fmaxf(v, 0.0f);
    if (RESID) v += Res[orow * N + col];
    if (OUTF) Cf[orow * N + col] = v;
    if (OUTB) Cb[orow * N + col] = (bf16_t)v;
  }
}

// ---------------------------------------------------------------------------
// conv0: (16,40000) -> relu -> avgpool4 -> c0 (16,8,10000)
// ---------------------------------------------------------------------------
__global__ void conv0_k(const float* __restrict__ in, const float* __restrict__ w,
                        const float* __restrict__ bias, float* __restrict__ c0) {
  int idx = blockIdx.x * 256 + threadIdx.x;
  if (idx >= 16 * 10000) return;
  int b = idx / 10000, t = idx % 10000;
  const float* xin = in + (size_t)b * 40000;
  float acc[8];
#pragma unroll
  for (int c = 0; c < 8; ++c) acc[c] = 0.0f;
  for (int i = 0; i < 4; ++i) {
    int base = t * 4 + i;
    float s[8];
#pragma unroll
    for (int c = 0; c < 8; ++c) s[c] = bias[c];
    for (int j = 0; j < 11; ++j) {
      int pos = base + j - 5;
      float xv = (pos >= 0 && pos < 40000) ? xin[pos] : 0.0f;
#pragma unroll
      for (int c = 0; c < 8; ++c) s[c] += w[c * 11 + j] * xv;
    }
#pragma unroll
    for (int c = 0; c < 8; ++c) acc[c] += fmaxf(s[c], 0.0f);
  }
#pragma unroll
  for (int c = 0; c < 8; ++c)
    c0[((size_t)b * 8 + c) * 10000 + t] = acc[c] * 0.25f;
}

// conv1: c0 (16,8,10000) -> relu -> avgpool4 -> transpose -> x (16,2500,128)
__global__ void conv1_k(const float* __restrict__ c0, const float* __restrict__ w,
                        const float* __restrict__ bias, float* __restrict__ x) {
  int idx = blockIdx.x * 256 + threadIdx.x;
  if (idx >= 16 * LCN * 128) return;
  int c = idx & 127;
  int bl = idx >> 7;
  int b = bl / LCN, l = bl % LCN;
  const float* src = c0 + (size_t)b * 8 * 10000;
  float acc = 0.0f;
  for (int i = 0; i < 4; ++i) {
    int t = l * 4 + i;
    float s = bias[c];
#pragma unroll
    for (int ci = 0; ci < 8; ++ci) {
      const float* row = src + ci * 10000;
#pragma unroll
      for (int j = 0; j < 3; ++j) {
        int pos = t + j - 1;
        if (pos >= 0 && pos < 10000) s += w[(c * 8 + ci) * 3 + j] * row[pos];
      }
    }
    acc += fmaxf(s, 0.0f);
  }
  x[(size_t)bl * 128 + c] = acc * 0.25f;
}

// ---------------------------------------------------------------------------
// layer_center: out_bf16 = x - mean(x,row) + bias   (one block per row)
// ---------------------------------------------------------------------------
__global__ void center_k(const float* __restrict__ x, const float* __restrict__ bias,
                         bf16_t* __restrict__ out) {
  __shared__ float red[128];
  int row = blockIdx.x;
  int t = threadIdx.x;
  float v = x[(size_t)row * 128 + t];
  red[t] = v;
  __syncthreads();
  for (int s = 64; s > 0; s >>= 1) {
    if (t < s) red[t] += red[t + s];
    __syncthreads();
  }
  float mean = red[0] * (1.0f / 128.0f);
  out[(size_t)row * 128 + t] = (bf16_t)(v - mean + bias[t]);
}

__global__ void cast_bf_k(const float* __restrict__ s, bf16_t* __restrict__ d, int n) {
  int i = blockIdx.x * 256 + threadIdx.x;
  if (i < n) d[i] = (bf16_t)s[i];
}

// ---------------------------------------------------------------------------
// z (40000x384 f32: [qp|kp|v]) -> qf[bh][l][32], kfT[bh][m][LKP], vbh[bh][l][16]
// ---------------------------------------------------------------------------
__global__ void reshape_qkv_k(const float* __restrict__ z, bf16_t* __restrict__ qf,
                              bf16_t* __restrict__ kfT, bf16_t* __restrict__ vbh) {
  int row = blockIdx.x;  // 0..39999
  int t = threadIdx.x;   // 0..127
  int b = row / LCN, l = row % LCN;
  int h = t >> 4, j = t & 15;
  int bh = b * 8 + h;
  float qp = z[(size_t)row * 384 + t];
  float kp = z[(size_t)row * 384 + 128 + t];
  float vv = z[(size_t)row * 384 + 256 + t];
  float sq, cq, sk, ck;
  __sincosf(qp, &sq, &cq);
  __sincosf(kp, &sk, &ck);
  const float r4 = 0.25f;  // 1/sqrt(D_KMAP)
  size_t qb = ((size_t)bh * LQP + l) * 32;
  qf[qb + j] = (bf16_t)(cq * r4);
  qf[qb + 16 + j] = (bf16_t)(sq * r4);
  size_t kb = ((size_t)bh * 32 + j) * LKP + l;
  kfT[kb] = (bf16_t)(ck * r4);
  kfT[kb + (size_t)16 * LKP] = (bf16_t)(sk * r4);
  vbh[((size_t)bh * LKP + l) * 16 + j] = (bf16_t)vv;
}

// zero the l-padding tails of qf / kfT / vbh (done once per call)
__global__ void pad_tails_k(bf16_t* __restrict__ qf, bf16_t* __restrict__ kfT,
                            bf16_t* __restrict__ vbh) {
  int tid = blockIdx.x * 256 + threadIdx.x;
  if (tid < NBH * 60 * 32) {  // qf rows [2500,2560)
    int m = tid & 31;
    int l = ((tid >> 5) % 60) + LCN;
    int bh = tid / (60 * 32);
    qf[((size_t)bh * LQP + l) * 32 + m] = (bf16_t)0.0f;
  }
  if (tid < NBH * 32 * 28) {  // kfT cols [2500,2528)
    int l = (tid % 28) + LCN;
    int m = (tid / 28) & 31;
    int bh = tid / (28 * 32);
    kfT[((size_t)bh * 32 + m) * LKP + l] = (bf16_t)0.0f;
  }
  if (tid < NBH * 28 * 16) {  // vbh rows [2500,2528)
    int d = tid & 15;
    int l = ((tid >> 4) % 28) + LCN;
    int bh = tid / (28 * 16);
    vbh[((size_t)bh * LKP + l) * 16 + d] = (bf16_t)0.0f;
  }
}

// ---------------------------------------------------------------------------
// Phase 1: S[b,h,p](32x16) = kfT(32 x L) * diag(kw[:,p]) * V(L x 16)
// Each wave owns the (p, p+32) forward/backward pair (same slope ns[p]):
// A fragments and the raw V fragment are loaded once and feed 4 WMMAs.
// grid (128, 4), 8 waves/block.
// ---------------------------------------------------------------------------
__global__ void phase1_S_k(const bf16_t* __restrict__ kfT,
                           const bf16_t* __restrict__ vbh,
                           const float* __restrict__ heads,
                           bf16_t* __restrict__ S) {
  const int bh = blockIdx.x;
  const int wave = threadIdx.x >> 5;
  const int lane = threadIdx.x & 31;
  const int p0 = blockIdx.y * 8 + wave;  // forward channel, 0..31
  const int p1 = p0 + 32;                // backward twin
  const int h = bh & 7;
  const float head = heads[h];
  const float ns = (4.0f - 0.125f * (float)p0) * (1.0f / 2499.0f);
  const float coef = -head * ns;  // kw_fwd = exp(coef*l), kw_bwd = exp(coef*(2499-l))

  v8f aF0 = wz8(), aF1 = wz8(), aB0 = wz8(), aB1 = wz8();
  const int mrow = lane & 15;
  const int lpart = (lane >> 4) * 8;
  const bf16_t* a0 = kfT + ((size_t)bh * 32 + mrow) * LKP;
  const bf16_t* a1 = a0 + (size_t)16 * LKP;
  const bf16_t* vb = vbh + (size_t)bh * LKP * 16;

  for (int kc = 0; kc < LKP; kc += 32) {
    const int lb = kc + lane;
    v16bf bF, bB;
#pragma unroll
    for (int e = 0; e < 16; ++e) {
      const int lk = kc + ((e < 8) ? (lpart + e) : (16 + lpart + e - 8));
      float t = (float)vb[(size_t)lk * 16 + (lane & 15)];
      bF[e] = (__bf16)(t * __expf(coef * (float)lk));
      bB[e] = (__bf16)(t * __expf(coef * (2499.0f - (float)lk)));
    }
    __builtin_prefetch(vb + (size_t)(lb + 32) * 16, 0, 1);

    v16bf af0 = join16(*(const v8bf*)(a0 + kc + lpart),
                       *(const v8bf*)(a0 + kc + 16 + lpart));
    v16bf af1 = join16(*(const v8bf*)(a1 + kc + lpart),
                       *(const v8bf*)(a1 + kc + 16 + lpart));
    aF0 = wmma_bf(af0, bF, aF0);
    aF1 = wmma_bf(af1, bF, aF1);
    aB0 = wmma_bf(af0, bB, aB0);
    aB1 = wmma_bf(af1, bB, aB1);
  }

  const int v = lane & 15;
  const int r8 = (lane >> 4) * 8;
  bf16_t* sp0 = S + ((size_t)bh * 64 + p0) * 512;
  bf16_t* sp1 = S + ((size_t)bh * 64 + p1) * 512;
#pragma unroll
  for (int r = 0; r < 8; ++r) {
    sp0[(r8 + r) * 16 + v] = (bf16_t)aF0[r];
    sp0[(16 + r8 + r) * 16 + v] = (bf16_t)aF1[r];
    sp1[(r8 + r) * 16 + v] = (bf16_t)aB0[r];
    sp1[(16 + r8 + r) * 16 + v] = (bf16_t)aB1[r];
  }
}

// ---------------------------------------------------------------------------
// Phase 2: attn_tile(16x16) = (1/L) * sum_p diag(qw[:,p]) * qf_tile * S[p]
// grid (128, 10); 8 waves share S[b,h] (64KB) staged in LDS (TDM when
// available), and each wave owns TWO l-tiles so every LDS B-fragment feeds
// two WMMAs.
// ---------------------------------------------------------------------------
__global__ void phase2_attn_k(const bf16_t* __restrict__ qf,
                              const bf16_t* __restrict__ S,
                              const float* __restrict__ heads,
                              bf16_t* __restrict__ attn) {
  extern __shared__ char smem[];
  bf16_t* sS = (bf16_t*)smem;  // 64*512 bf16 = 64KB

  const int bh = blockIdx.x;
  const bf16_t* sg = S + (size_t)bh * 64 * 512;

#if defined(__HIP_DEVICE_COMPILE__) && \
    __has_builtin(__builtin_amdgcn_tensor_load_to_lds) && \
    __has_builtin(__builtin_amdgcn_s_wait_tensorcnt)
  // Tensor Data Mover: one wave issues a 64KB contiguous-tile DMA into LDS.
  if (threadIdx.x < 32) {
    const unsigned ldsoff = (unsigned)(uintptr_t)sS;  // addr[31:0] = LDS byte offset
    const unsigned long long ga = (unsigned long long)(uintptr_t)sg;
    u32x4 g0;
    g0[0] = 1u;                                   // count=1, user descriptor
    g0[1] = ldsoff;                               // lds_addr
    g0[2] = (unsigned)(ga & 0xffffffffu);         // global_addr[31:0]
    g0[3] = (unsigned)((ga >> 32) & 0x1ffffffu)   // global_addr[56:32]
            | (2u << 30);                         // type = 2 ("image")
    i32x8 g1;
    g1[0] = 0x30000;     // data_size = 3 (8B); no multicast / barrier / pad
    g1[1] = 0x20000000;  // tensor_dim0 = 8192 (8B units) in bits [79:48] -> low part
    g1[2] = 0x10000;     // tensor_dim0 hi = 0, tensor_dim1 = 1
    g1[3] = 0x20000000;  // tile_dim0 = 8192 in bits [127:112]
    g1[4] = 1;           // tile_dim1 = 1
    g1[5] = 8192;        // tensor_dim0_stride (8B units), bits [207:160]
    g1[6] = 0;
    g1[7] = 0;
    i32x4 z4;
    z4[0] = z4[1] = z4[2] = z4[3] = 0;
#if __clang_major__ >= 23
    i32x8 z8;
#pragma unroll
    for (int i = 0; i < 8; ++i) z8[i] = 0;
    __builtin_amdgcn_tensor_load_to_lds(g0, g1, z4, z4, z8, 0);
#else
    __builtin_amdgcn_tensor_load_to_lds(g0, g1, z4, z4, 0);
#endif
    __builtin_amdgcn_s_wait_tensorcnt(0);
  }
  __syncthreads();
#else
  {
    const uint4* src = (const uint4*)sg;
    uint4* dst = (uint4*)sS;
    for (int i = threadIdx.x; i < 4096; i += 256) dst[i] = src[i];
    __syncthreads();
  }
#endif

  const int wave = threadIdx.x >> 5;
  const int lane = threadIdx.x & 31;
  const int h = bh & 7;
  const float head = heads[h];
  const int tile0 = blockIdx.y * 8 + wave;  // 0..79
  const int tile1 = tile0 + 80;             // 80..159 (padded rows are zero)
  const int lr0 = tile0 * 16 + (lane & 15);
  const int lr1 = tile1 * 16 + (lane & 15);
  const int mpart = (lane >> 4) * 8;

  const bf16_t* ab0 = qf + ((size_t)bh * LQP + lr0) * 32;
  const bf16_t* ab1 = qf + ((size_t)bh * LQP + lr1) * 32;
  v16bf a0 = join16(*(const v8bf*)(ab0 + mpart), *(const v8bf*)(ab0 + 16 + mpart));
  v16bf a1 = join16(*(const v8bf*)(ab1 + mpart), *(const v8bf*)(ab1 + 16 + mpart));

  v8f acc0 = wz8(), acc1 = wz8();
  for (int p = 0; p < 64; ++p) {
    const int jj = p & 31;
    const float ns = (4.0f - 0.125f * (float)jj) * (1.0f / 2499.0f);
    const float hns = head * ns;
    const float qw0 = __expf(hns * ((p < 32) ? (float)lr0 : (2499.0f - (float)lr0)));
    const float qw1 = __expf(hns * ((p < 32) ? (float)lr1 : (2499.0f - (float)lr1)));
    v16bf as0, as1;
#pragma unroll
    for (int e = 0; e < 16; ++e) {
      as0[e] = (__bf16)((float)a0[e] * qw0);
      as1[e] = (__bf16)((float)a1[e] * qw1);
    }
    v16bf b;
#pragma unroll
    for (int e = 0; e < 16; ++e)
      b[e] = sS[p * 512 + ((e < 8) ? (mpart + e) : (16 + mpart + e - 8)) * 16 + (lane & 15)];
    acc0 = wmma_bf(as0, b, acc0);
    acc1 = wmma_bf(as1, b, acc1);
  }

  const int b_ = bh >> 3;
  const int col = h * 16 + (lane & 15);
  const float invl = 1.0f / (float)LCN;
  const int rb0 = tile0 * 16 + (lane >> 4) * 8;
  const int rb1 = tile1 * 16 + (lane >> 4) * 8;
#pragma unroll
  for (int r = 0; r < 8; ++r) {
    int row0 = rb0 + r;
    if (row0 < LCN)
      attn[((size_t)b_ * LCN + row0) * 128 + col] = (bf16_t)(acc0[r] * invl);
    int row1 = rb1 + r;
    if (row1 < LCN)
      attn[((size_t)b_ * LCN + row1) * 128 + col] = (bf16_t)(acc1[r] * invl);
  }
}

// ---------------------------------------------------------------------------
// Weight preparation
// ---------------------------------------------------------------------------
// W_all (128 x 384 bf16) = [ inv_sc*Wq@omega | inv_sc*Wk@omega | Wv ]
__global__ void fold_omega_k(const float* __restrict__ Wq, const float* __restrict__ Wk,
                             const float* __restrict__ Wv, const float* __restrict__ om,
                             bf16_t* __restrict__ Wall) {
  int idx = blockIdx.x * 256 + threadIdx.x;
  if (idx >= 128 * 384) return;
  int r = idx / 384, c = idx % 384;
  float out;
  if (c < 256) {
    const float* W = (c < 128) ? Wq : Wk;
    int cc = c & 127, h = cc >> 4, m = cc & 15;
    float s = 0.0f;
#pragma unroll
    for (int d = 0; d < 16; ++d)
      s += W[r * 128 + h * 16 + d] * om[(h * 16 + d) * 16 + m];
    out = 0.5f * s;  // inv_sc = 16^-0.25
  } else {
    out = Wv[r * 128 + (c - 256)];
  }
  Wall[(size_t)r * 384 + c] = (bf16_t)out;
}

__global__ void fold_bias_k(const float* __restrict__ bq, const float* __restrict__ bk,
                            const float* __restrict__ bv, const float* __restrict__ om,
                            float* __restrict__ ball) {
  int c = blockIdx.x * 256 + threadIdx.x;
  if (c >= 384) return;
  float out;
  if (c < 256) {
    const float* bb = (c < 128) ? bq : bk;
    int cc = c & 127, h = cc >> 4, m = cc & 15;
    float s = 0.0f;
#pragma unroll
    for (int d = 0; d < 16; ++d) s += bb[h * 16 + d] * om[(h * 16 + d) * 16 + m];
    out = 0.5f * s;
  } else {
    out = bv[c - 256];
  }
  ball[c] = out;
}

__global__ void pack2_bf_k(const float* __restrict__ a, const float* __restrict__ b,
                           bf16_t* __restrict__ d, int rows, int cols) {
  int i = blockIdx.x * 256 + threadIdx.x;
  if (i >= rows * cols * 2) return;
  int r = i / (2 * cols), c = i % (2 * cols);
  float v = (c < cols) ? a[r * cols + c] : b[r * cols + (c - cols)];
  d[i] = (bf16_t)v;
}

__global__ void pack2_f32_k(const float* __restrict__ a, const float* __restrict__ b,
                            float* __restrict__ d, int n) {
  int i = blockIdx.x * 256 + threadIdx.x;
  if (i >= 2 * n) return;
  d[i] = (i < n) ? a[i] : b[i - n];
}

// ---------------------------------------------------------------------------
// Head: per (b,h) softmax over n=0..2499 of score, pooled[d] = sum prob*vh
// kv layout: (40000 x 256 f32) cols [0,128)=kh (d*8+h), [128,256)=vh.
// ---------------------------------------------------------------------------
__global__ void softmax_pool_k(const float* __restrict__ kv, const float* __restrict__ qh,
                               bf16_t* __restrict__ pooled_b, float* __restrict__ sws) {
  __shared__ float red[256];
  const int b = blockIdx.x >> 3, h = blockIdx.x & 7;
  const int t = threadIdx.x;
  const float sc = 0.25f;  // D_HEAD^-0.5
  float* score = sws + (size_t)blockIdx.x * LCN;

  float lmax = -1e30f;
  for (int n = t; n < LCN; n += 256) {
    const float* rowp = kv + ((size_t)b * LCN + n) * 256;
    float s = 0.0f;
#pragma unroll
    for (int d = 0; d < 16; ++d) s += rowp[d * 8 + h] * qh[d * 8 + h];
    s *= sc;
    score[n] = s;
    lmax = fmaxf(lmax, s);
  }
  red[t] = lmax;
  __syncthreads();
  for (int s2 = 128; s2 > 0; s2 >>= 1) {
    if (t < s2) red[t] = fmaxf(red[t], red[t + s2]);
    __syncthreads();
  }
  const float gmax = red[0];
  __syncthreads();

  float lsum = 0.0f;
  for (int n = t; n < LCN; n += 256) lsum += __expf(score[n] - gmax);
  red[t] = lsum;
  __syncthreads();
  for (int s2 = 128; s2 > 0; s2 >>= 1) {
    if (t < s2) red[t] += red[t + s2];
    __syncthreads();
  }
  const float inv = 1.0f / red[0];
  __syncthreads();

  float pacc[16];
#pragma unroll
  for (int d = 0; d < 16; ++d) pacc[d] = 0.0f;
  for (int n = t; n < LCN; n += 256) {
    float pr = __expf(score[n] - gmax) * inv;
    const float* rowp = kv + ((size_t)b * LCN + n) * 256 + 128;
#pragma unroll
    for (int d = 0; d < 16; ++d) pacc[d] += rowp[d * 8 + h] * pr;
  }
  for (int d = 0; d < 16; ++d) {
    red[t] = pacc[d];
    __syncthreads();
    for (int s2 = 128; s2 > 0; s2 >>= 1) {
      if (t < s2) red[t] += red[t + s2];
      __syncthreads();
    }
    if (t == 0) pooled_b[(size_t)b * 128 + h * 16 + d] = (bf16_t)red[0];
    __syncthreads();
  }
}

// ---------------------------------------------------------------------------
// Host-side orchestration
// ---------------------------------------------------------------------------
extern "C" void kernel_launch(void* const* d_in, const int* in_sizes, int n_in,
                              void* d_out, int out_size, void* d_ws, size_t ws_size,
                              hipStream_t stream) {
  (void)in_sizes; (void)n_in; (void)out_size; (void)ws_size;
  const float* inp    = (const float*)d_in[0];
  const float* c0w    = (const float*)d_in[1];
  const float* c0b    = (const float*)d_in[2];
  const float* c1w    = (const float*)d_in[3];
  const float* c1b    = (const float*)d_in[4];
  const float* smq    = (const float*)d_in[37];
  const float* smkw   = (const float*)d_in[38];
  const float* smkb   = (const float*)d_in[39];
  const float* smvw   = (const float*)d_in[40];
  const float* smvb   = (const float*)d_in[41];
  const float* fcw    = (const float*)d_in[42];
  const float* fcb    = (const float*)d_in[43];

  char* ws = (char*)d_ws;
  size_t off = 0;
  auto alloc = [&](size_t bytes) -> void* {
    void* p = ws + off;
    off = (off + bytes + 255) & ~(size_t)255;
    return p;
  };

  float*  c0     = (float*) alloc((size_t)16 * 8 * 10000 * 4);
  float*  x      = (float*) alloc((size_t)40000 * 128 * 4);
  bf16_t* xb     = (bf16_t*)alloc((size_t)40000 * 128 * 2);
  float*  zbuf   = (float*) alloc((size_t)40000 * 384 * 4);  // reused: kv, hbuf, score
  bf16_t* qf     = (bf16_t*)alloc((size_t)NBH * LQP * 32 * 2);
  bf16_t* kfT    = (bf16_t*)alloc((size_t)NBH * 32 * LKP * 2);
  bf16_t* vbh    = (bf16_t*)alloc((size_t)NBH * LKP * 16 * 2);
  bf16_t* Sbf    = (bf16_t*)alloc((size_t)NBH * 64 * 512 * 2);
  bf16_t* attnb  = (bf16_t*)alloc((size_t)40000 * 128 * 2);
  bf16_t* Wall   = (bf16_t*)alloc((size_t)2 * 128 * 384 * 2);
  float*  ball   = (float*) alloc((size_t)2 * 384 * 4);
  bf16_t* Wobf   = (bf16_t*)alloc((size_t)2 * 16384 * 2);
  bf16_t* ff1bf  = (bf16_t*)alloc((size_t)2 * 128 * 256 * 2);
  bf16_t* ff2bf  = (bf16_t*)alloc((size_t)2 * 256 * 128 * 2);
  bf16_t* smkvbf = (bf16_t*)alloc((size_t)128 * 256 * 2);
  float*  bkv    = (float*) alloc((size_t)256 * 4);
  bf16_t* fcbf   = (bf16_t*)alloc((size_t)128 * 256 * 2);
  bf16_t* pooledb= (bf16_t*)alloc((size_t)16 * 128 * 2);

  bf16_t* hbuf   = (bf16_t*)zbuf;                       // FF inner (40000x256 bf16)
  float*  kv     = zbuf;                                // head kv (40000x256 f32)
  float*  sws    = zbuf + (size_t)40000 * 256;          // score scratch (128x2500)

  // ---- weight prep ----
  for (int i = 0; i < 2; ++i) {
    const int lb = 5 + 16 * i;
    const float* Wq = (const float*)d_in[lb + 0];
    const float* Wk = (const float*)d_in[lb + 1];
    const float* Wv = (const float*)d_in[lb + 2];
    const float* Wo = (const float*)d_in[lb + 3];
    const float* bq = (const float*)d_in[lb + 4];
    const float* bk = (const float*)d_in[lb + 5];
    const float* bv = (const float*)d_in[lb + 6];
    const float* om = (const float*)d_in[lb + 8];
    const float* f1 = (const float*)d_in[lb + 10];
    const float* f2 = (const float*)d_in[lb + 12];
    fold_omega_k<<<(128 * 384 + 255) / 256, 256, 0, stream>>>(Wq, Wk, Wv, om,
                                                              Wall + (size_t)i * 128 * 384);
    fold_bias_k<<<2, 256, 0, stream>>>(bq, bk, bv, om, ball + i * 384);
    cast_bf_k<<<(16384 + 255) / 256, 256, 0, stream>>>(Wo, Wobf + (size_t)i * 16384, 16384);
    cast_bf_k<<<(32768 + 255) / 256, 256, 0, stream>>>(f1, ff1bf + (size_t)i * 32768, 32768);
    cast_bf_k<<<(32768 + 255) / 256, 256, 0, stream>>>(f2, ff2bf + (size_t)i * 32768, 32768);
  }
  pack2_bf_k<<<(128 * 256 + 255) / 256, 256, 0, stream>>>(smkw, smvw, smkvbf, 128, 128);
  pack2_f32_k<<<1, 256, 0, stream>>>(smkb, smvb, bkv, 128);
  cast_bf_k<<<(32768 + 255) / 256, 256, 0, stream>>>(fcw, fcbf, 32768);
  pad_tails_k<<<(NBH * 60 * 32 + 255) / 256, 256, 0, stream>>>(qf, kfT, vbh);

  // ---- conv front-end ----
  conv0_k<<<(16 * 10000 + 255) / 256, 256, 0, stream>>>(inp, c0w, c0b, c0);
  conv1_k<<<(16 * LCN * 128 + 255) / 256, 256, 0, stream>>>(c0, c1w, c1b, x);

  // ---- transformer layers ----
  for (int i = 0; i < 2; ++i) {
    const int lb = 5 + 16 * i;
    const float* bo    = (const float*)d_in[lb + 7];
    const float* heads = (const float*)d_in[lb + 9];
    const float* f1b   = (const float*)d_in[lb + 11];
    const float* f2b   = (const float*)d_in[lb + 13];
    const float* lc1b  = (const float*)d_in[lb + 14];
    const float* lc2b  = (const float*)d_in[lb + 15];

    center_k<<<40000, 128, 0, stream>>>(x, lc1b, xb);
    gemm_wmma_k<false, false, true, false><<<dim3(2500, 3), 256, 0, stream>>>(
        xb, Wall + (size_t)i * 128 * 384, ball + i * 384, zbuf, nullptr, nullptr, 384, 128);
    reshape_qkv_k<<<40000, 128, 0, stream>>>(zbuf, qf, kfT, vbh);
    phase1_S_k<<<dim3(NBH, 4), 256, 0, stream>>>(kfT, vbh, heads, Sbf);
    phase2_attn_k<<<dim3(NBH, 10), 256, 65536, stream>>>(qf, Sbf, heads, attnb);
    gemm_wmma_k<false, true, true, false><<<dim3(2500, 1), 256, 0, stream>>>(
        attnb, Wobf + (size_t)i * 16384, bo, x, nullptr, x, 128, 128);

    center_k<<<40000, 128, 0, stream>>>(x, lc2b, xb);
    gemm_wmma_k<true, false, false, true><<<dim3(2500, 2), 256, 0, stream>>>(
        xb, ff1bf + (size_t)i * 32768, f1b, nullptr, hbuf, nullptr, 256, 128);
    gemm_wmma_k<false, true, true, false><<<dim3(2500, 1), 256, 0, stream>>>(
        hbuf, ff2bf + (size_t)i * 32768, f2b, x, nullptr, x, 128, 256);
  }

  // ---- attention-pooling head ----
  cast_bf_k<<<(40000 * 128 + 255) / 256, 256, 0, stream>>>(x, xb, 40000 * 128);
  gemm_wmma_k<false, false, true, false><<<dim3(2500, 2), 256, 0, stream>>>(
      xb, smkvbf, bkv, kv, nullptr, nullptr, 256, 128);
  softmax_pool_k<<<NBH, 256, 0, stream>>>(kv, smq, pooledb, sws);
  gemm_wmma_k<false, false, true, false><<<dim3(1, 2), 256, 0, stream>>>(
      pooledb, fcbf, fcb, (float*)d_out, nullptr, nullptr, 256, 128);
}
